// FusedMoE_88141318848692
// MI455X (gfx1250) — matrix operand order repaired and checked by reading, not compiled, not run on hardware
//
#include <hip/hip_runtime.h>
#include <cstdint>
#include <cstddef>

typedef __attribute__((ext_vector_type(16))) __bf16          v16bf;
typedef __attribute__((ext_vector_type(8)))  float           v8f;
typedef __attribute__((ext_vector_type(8)))  unsigned short  v8u16;

namespace {
constexpr int T_TOK = 4096;
constexpr int NEXP  = 8;
constexpr int HDIM  = 1024;
constexpr int IDIM  = 2048;
constexpr int TOPK  = 2;

constexpr int MT  = 64;        // block M tile (token-slots)
constexpr int NT  = 128;       // block N tile
constexpr int KT  = 32;        // K step (one bf16 WMMA depth)
constexpr int SKP = KT + 8;    // padded LDS row stride in ushort (80B rows)

constexpr int SLOT_CAP = 8704; // 2*T + E*(MT-1), rounded up

constexpr size_t align256(size_t x){ return (x + 255) & ~size_t(255); }
constexpr size_t OFF_TOPK_ID = 0;
constexpr size_t OFF_TOPK_W  = align256(OFF_TOPK_ID + sizeof(int)   * T_TOK * TOPK);
constexpr size_t OFF_CNT     = align256(OFF_TOPK_W  + sizeof(float) * T_TOK * TOPK);
constexpr size_t OFF_PADOFF  = align256(OFF_CNT     + sizeof(int)   * NEXP);
constexpr size_t OFF_POS     = align256(OFF_PADOFF  + sizeof(int)   * (NEXP + 1));
constexpr size_t OFF_LIST    = align256(OFF_POS     + sizeof(int)   * T_TOK * TOPK);
constexpr size_t OFF_ACT     = align256(OFF_LIST    + sizeof(int)   * SLOT_CAP);
constexpr size_t OFF_Y       = align256(OFF_ACT   + sizeof(unsigned short) * (size_t)SLOT_CAP * IDIM);
constexpr size_t OFF_XBF     = align256(OFF_Y     + sizeof(float)          * (size_t)SLOT_CAP * HDIM);
constexpr size_t OFF_W13BF   = align256(OFF_XBF   + sizeof(unsigned short) * (size_t)T_TOK * HDIM);
constexpr size_t OFF_W2BF    = align256(OFF_W13BF + sizeof(unsigned short) * (size_t)NEXP * 2 * IDIM * HDIM);
} // namespace

__device__ __forceinline__ unsigned short f2bf(float f){
  union { float f; unsigned u; } c; c.f = f;
  unsigned u = c.u + 0x7fffu + ((c.u >> 16) & 1u);   // round-to-nearest-even
  return (unsigned short)(u >> 16);
}

__device__ __forceinline__ v8u16 pack8(float4 a, float4 b){
  v8u16 r;
  r[0]=f2bf(a.x); r[1]=f2bf(a.y); r[2]=f2bf(a.z); r[3]=f2bf(a.w);
  r[4]=f2bf(b.x); r[5]=f2bf(b.y); r[6]=f2bf(b.z); r[7]=f2bf(b.w);
  return r;
}

__device__ __forceinline__ v16bf cat16(v8u16 lo, v8u16 hi){
  union { v16bf b; v8u16 u[2]; } t;
  t.u[0] = lo; t.u[1] = hi;
  return t.b;
}

__device__ __forceinline__ v8f wmma_bf16(v16bf a, v16bf b, v8f c){
  return __builtin_amdgcn_wmma_f32_16x16x32_bf16(false, a, false, b, (short)0, c, false, false);
}

// Async 16B global -> LDS copy (per-lane, ASYNCcnt-tracked).
__device__ __forceinline__ void async_cp16(unsigned short* lds, const unsigned short* gsrc){
  asm volatile("global_load_async_to_lds_b128 %0, %1, off"
               :: "v"((unsigned)(unsigned long long)(void*)lds), "v"(gsrc)
               : "memory");
}
__device__ __forceinline__ void wait_async0(){
  asm volatile("s_wait_asynccnt 0x0" ::: "memory");
}

// ------------------------------------------------- fp32 -> bf16 pre-pass ----
__global__ __launch_bounds__(256)
void moe_cvt_bf16(const float* __restrict__ src, unsigned short* __restrict__ dst, long n8){
  long i = (long)blockIdx.x * blockDim.x + threadIdx.x;
  if (i >= n8) return;
  const float* p = src + i * 8;
  float4 a = *(const float4*)p;
  float4 b = *(const float4*)(p + 4);
  __builtin_prefetch(p + 2048, 0, 0);
  *(v8u16*)(dst + i * 8) = pack8(a, b);
}

// ---------------------------------------------------------------- routing ---
__global__ void moe_zero_cnt(int* cnt){
  if (threadIdx.x < NEXP) cnt[threadIdx.x] = 0;
}

__global__ void moe_route(const float* __restrict__ logits, int* __restrict__ topk_id,
                          float* __restrict__ topk_w, int* __restrict__ cnt){
  int t = blockIdx.x * blockDim.x + threadIdx.x;
  if (t >= T_TOK) return;
  float l[NEXP];
  #pragma unroll
  for (int j = 0; j < NEXP; ++j) l[j] = logits[t * NEXP + j];
  int i0 = 0; float b0 = l[0];
  #pragma unroll
  for (int j = 1; j < NEXP; ++j) if (l[j] > b0){ b0 = l[j]; i0 = j; }
  int i1 = (i0 == 0) ? 1 : 0; float b1 = l[i1];
  #pragma unroll
  for (int j = 0; j < NEXP; ++j) if (j != i0 && l[j] > b1){ b1 = l[j]; i1 = j; }
  // softmax denominator cancels under top-2 renormalization
  float e1 = __expf(b1 - b0);
  float w0 = 1.f / (1.f + e1);
  topk_id[2*t]   = i0;  topk_id[2*t+1] = i1;
  topk_w [2*t]   = w0;  topk_w [2*t+1] = 1.f - w0;
  atomicAdd(&cnt[i0], 1);
  atomicAdd(&cnt[i1], 1);
}

__global__ void moe_scan(const int* __restrict__ cnt, int* __restrict__ padoff){
  if (threadIdx.x == 0){
    int acc = 0;
    for (int e = 0; e < NEXP; ++e){
      padoff[e] = acc;
      acc += (cnt[e] + MT - 1) / MT * MT;
    }
    padoff[NEXP] = acc;
  }
}

__global__ void moe_compact(const int* __restrict__ topk_id, const int* __restrict__ padoff,
                            int* __restrict__ list, int* __restrict__ pos){
  __shared__ int s[256];
  const int e = blockIdx.x, tid = threadIdx.x;
  const int base_out = padoff[e];
  int running = 0;
  for (int base = 0; base < T_TOK; base += 256){
    int t = base + tid;
    int k = -1;
    if      (topk_id[2*t]   == e) k = 0;
    else if (topk_id[2*t+1] == e) k = 1;
    int f = (k >= 0) ? 1 : 0;
    s[tid] = f; __syncthreads();
    for (int off = 1; off < 256; off <<= 1){
      int v = (tid >= off) ? s[tid - off] : 0;
      __syncthreads();
      s[tid] += v;
      __syncthreads();
    }
    int incl = s[tid], tot = s[255];
    if (f){
      int p = running + incl - 1;        // stable: ordered by token index
      list[base_out + p] = t;
      pos[2*t + k] = p;
    }
    running += tot;
    __syncthreads();
  }
}

// ------------------------------------------------------- GEMM1: gate/up -----
// act[slot, i] = silu(x@Wg^T) * (x@Wu^T), bf16 out. Block tile 64 x (128g+128u).
// Double-buffered async-to-LDS pipeline; all LDS data already bf16.
__global__ __launch_bounds__(256)
void moe_gemm1(const unsigned short* __restrict__ xbf, const unsigned short* __restrict__ w13bf,
               const int* __restrict__ cnt, const int* __restrict__ padoff,
               const int* __restrict__ list, unsigned short* __restrict__ act){
  __shared__ unsigned short sA [2][MT][SKP];
  __shared__ unsigned short sBg[2][NT][SKP];
  __shared__ unsigned short sBu[2][NT][SKP];

  const int e    = blockIdx.z;
  const int cnte = cnt[e];
  const int m0   = blockIdx.y * MT;
  if (m0 >= cnte) return;                 // uniform early-exit (before barriers)
  const int nblk = blockIdx.x;
  const long slotBase = (long)padoff[e] + m0;

  const int tid  = threadIdx.x;
  const int lane = tid & 31, wave = tid >> 5;
  const int wm   = wave & 1,  wn  = wave >> 1;   // 2 M-waves x 4 N-waves
  const int half = lane >> 4, l15 = lane & 15;

  const int ar = tid >> 2, ac = (tid & 3) * 8;   // A stager: 64 rows x 32k, 16B each
  const int br = tid >> 1, bc = (tid & 1) * 16;  // B stager: 128 rows x 32k, 32B each

  // clamp invalid rows to tile row 0 (always valid since m0 < cnte);
  // their garbage products land only in output rows masked at the epilogue.
  const int arEff = (m0 + ar < cnte) ? ar : 0;
  const int tok   = list[slotBase + arEff];
  const unsigned short* xrow = xbf + (size_t)tok * HDIM;
  const unsigned short* gr   = w13bf + (size_t)(e * 2 * IDIM + nblk * NT + br) * HDIM;
  const unsigned short* ur   = gr + (size_t)IDIM * HDIM;

  v8f accg[2][2] = {};
  v8f accu[2][2] = {};

  auto issue = [&](int kk, int buf){
    async_cp16(&sA [buf][ar][ac],     xrow + kk + ac);
    async_cp16(&sBg[buf][br][bc],     gr + kk + bc);
    async_cp16(&sBg[buf][br][bc + 8], gr + kk + bc + 8);
    async_cp16(&sBu[buf][br][bc],     ur + kk + bc);
    async_cp16(&sBu[buf][br][bc + 8], ur + kk + bc + 8);
  };

  constexpr int NSTEP = HDIM / KT;
  issue(0, 0);
  for (int s = 0; s < NSTEP; ++s){
    wait_async0();                       // my tile-s copies have landed
    __syncthreads();                     // everyone's tile-s copies have landed
    const int buf = s & 1;
    if (s + 1 < NSTEP) issue((s + 1) * KT, buf ^ 1);

    v16bf afr[2];
    #pragma unroll
    for (int mi = 0; mi < 2; ++mi){
      const int row = wm*32 + mi*16 + l15;
      // A lanes<16: K 0..15 ; lanes>=16: K 16..31
      afr[mi] = cat16(*(const v8u16*)&sA[buf][row][half * 16],
                      *(const v8u16*)&sA[buf][row][half * 16 + 8]);
    }
    #pragma unroll
    for (int ni = 0; ni < 2; ++ni){
      const int col = wn*32 + ni*16 + l15;
      const int kb  = half * 16;         // B lanes<16: K 0..15 ; lanes>=16: K 16..31
      v16bf bg = cat16(*(const v8u16*)&sBg[buf][col][kb], *(const v8u16*)&sBg[buf][col][kb + 8]);
      v16bf bu = cat16(*(const v8u16*)&sBu[buf][col][kb], *(const v8u16*)&sBu[buf][col][kb + 8]);
      #pragma unroll
      for (int mi = 0; mi < 2; ++mi){
        accg[mi][ni] = wmma_bf16(afr[mi], bg, accg[mi][ni]);
        accu[mi][ni] = wmma_bf16(afr[mi], bu, accu[mi][ni]);
      }
    }
  }

  // fused SiLU(gate) * up epilogue -> bf16 activations
  #pragma unroll
  for (int mi = 0; mi < 2; ++mi){
    #pragma unroll
    for (int r = 0; r < 8; ++r){
      const int mloc = wm*32 + mi*16 + half*8 + r;  // C layout: VGPR r -> M = r + 8*(lane>=16)
      if (m0 + mloc < cnte){
        unsigned short* arow = act + (size_t)(slotBase + mloc) * IDIM;
        #pragma unroll
        for (int ni = 0; ni < 2; ++ni){
          const int n = nblk*NT + wn*32 + ni*16 + l15;
          float g = accg[mi][ni][r];
          float u = accu[mi][ni][r];
          float a = g * u / (1.f + __expf(-g));
          arow[n] = f2bf(a);
        }
      }
    }
  }
}

// ------------------------------------------------------- GEMM2: down proj ---
__global__ __launch_bounds__(256)
void moe_gemm2(const unsigned short* __restrict__ act, const unsigned short* __restrict__ w2bf,
               const int* __restrict__ cnt, const int* __restrict__ padoff,
               float* __restrict__ y){
  __shared__ unsigned short sA[2][MT][SKP];
  __shared__ unsigned short sB[2][NT][SKP];

  const int e    = blockIdx.z;
  const int cnte = cnt[e];
  const int m0   = blockIdx.y * MT;
  if (m0 >= cnte) return;
  const int nblk = blockIdx.x;
  const long slotBase = (long)padoff[e] + m0;

  const int tid  = threadIdx.x;
  const int lane = tid & 31, wave = tid >> 5;
  const int wm   = wave & 1,  wn  = wave >> 1;
  const int half = lane >> 4, l15 = lane & 15;

  const int ar = tid >> 2, ac = (tid & 3) * 8;
  const int br = tid >> 1, bc = (tid & 1) * 16;

  // padded act rows are real memory (finite garbage); outputs masked at store.
  const unsigned short* aptr = act + (size_t)(slotBase + ar) * IDIM;
  const unsigned short* brow = w2bf + ((size_t)e * HDIM + nblk * NT + br) * IDIM;

  v8f acc[2][2] = {};

  auto issue = [&](int kk, int buf){
    async_cp16(&sA[buf][ar][ac],     aptr + kk + ac);
    async_cp16(&sB[buf][br][bc],     brow + kk + bc);
    async_cp16(&sB[buf][br][bc + 8], brow + kk + bc + 8);
  };

  constexpr int NSTEP = IDIM / KT;
  issue(0, 0);
  for (int s = 0; s < NSTEP; ++s){
    wait_async0();
    __syncthreads();
    const int buf = s & 1;
    if (s + 1 < NSTEP) issue((s + 1) * KT, buf ^ 1);

    v16bf afr[2];
    #pragma unroll
    for (int mi = 0; mi < 2; ++mi){
      const int row = wm*32 + mi*16 + l15;
      afr[mi] = cat16(*(const v8u16*)&sA[buf][row][half * 16],
                      *(const v8u16*)&sA[buf][row][half * 16 + 8]);
    }
    #pragma unroll
    for (int ni = 0; ni < 2; ++ni){
      const int col = wn*32 + ni*16 + l15;
      const int kb  = half * 16;
      v16bf bf = cat16(*(const v8u16*)&sB[buf][col][kb], *(const v8u16*)&sB[buf][col][kb + 8]);
      #pragma unroll
      for (int mi = 0; mi < 2; ++mi)
        acc[mi][ni] = wmma_bf16(afr[mi], bf, acc[mi][ni]);
    }
  }

  #pragma unroll
  for (int mi = 0; mi < 2; ++mi){
    #pragma unroll
    for (int r = 0; r < 8; ++r){
      const int mloc = wm*32 + mi*16 + half*8 + r;
      if (m0 + mloc < cnte){
        float* yrow = y + (size_t)(slotBase + mloc) * HDIM;
        #pragma unroll
        for (int ni = 0; ni < 2; ++ni){
          const int n = nblk*NT + wn*32 + ni*16 + l15;
          yrow[n] = acc[mi][ni][r];
        }
      }
    }
  }
}

// ----------------------------------------------------------- combine --------
__global__ void moe_combine(const float* __restrict__ y, const int* __restrict__ topk_id,
                            const float* __restrict__ topk_w, const int* __restrict__ pos,
                            const int* __restrict__ padoff, float* __restrict__ out){
  int idx = blockIdx.x * blockDim.x + threadIdx.x;   // [0, T*H/4)
  int t = idx >> 8;                                  // H/4 == 256
  int h = (idx & 255) * 4;
  int   i0 = topk_id[2*t],   i1 = topk_id[2*t+1];
  float w0 = topk_w[2*t],    w1 = topk_w[2*t+1];
  long  s0 = (long)padoff[i0] + pos[2*t];
  long  s1 = (long)padoff[i1] + pos[2*t+1];
  const float4 a = *(const float4*)(y + (size_t)s0 * HDIM + h);
  const float4 b = *(const float4*)(y + (size_t)s1 * HDIM + h);
  float4 o;
  o.x = w0*a.x + w1*b.x;  o.y = w0*a.y + w1*b.y;
  o.z = w0*a.z + w1*b.z;  o.w = w0*a.w + w1*b.w;
  *(float4*)(out + (size_t)t * HDIM + h) = o;
}

// ----------------------------------------------------------- launcher -------
extern "C" void kernel_launch(void* const* d_in, const int* in_sizes, int n_in,
                              void* d_out, int out_size, void* d_ws, size_t ws_size,
                              hipStream_t stream){
  const float* x      = (const float*)d_in[0];
  const float* logits = (const float*)d_in[1];
  const float* w13    = (const float*)d_in[2];
  const float* w2     = (const float*)d_in[3];
  // d_in[4] = top_k (hardcoded to 2)
  float* out = (float*)d_out;

  char* ws = (char*)d_ws;
  int*            topk_id = (int*)  (ws + OFF_TOPK_ID);
  float*          topk_w  = (float*)(ws + OFF_TOPK_W);
  int*            cnt     = (int*)  (ws + OFF_CNT);
  int*            padoff  = (int*)  (ws + OFF_PADOFF);
  int*            pos     = (int*)  (ws + OFF_POS);
  int*            list    = (int*)  (ws + OFF_LIST);
  unsigned short* act     = (unsigned short*)(ws + OFF_ACT);
  float*          y       = (float*)(ws + OFF_Y);
  unsigned short* xbf     = (unsigned short*)(ws + OFF_XBF);
  unsigned short* w13bf   = (unsigned short*)(ws + OFF_W13BF);
  unsigned short* w2bf    = (unsigned short*)(ws + OFF_W2BF);

  const long n8_x   = (long)T_TOK * HDIM / 8;              //  524288
  const long n8_w13 = (long)NEXP * 2 * IDIM * HDIM / 8;    // 4194304
  const long n8_w2  = (long)NEXP * HDIM * IDIM / 8;        // 2097152

  moe_zero_cnt<<<1, 32, 0, stream>>>(cnt);
  moe_route  <<<T_TOK / 256, 256, 0, stream>>>(logits, topk_id, topk_w, cnt);
  moe_scan   <<<1, 32, 0, stream>>>(cnt, padoff);
  moe_compact<<<NEXP, 256, 0, stream>>>(topk_id, padoff, list, pos);

  moe_cvt_bf16<<<(int)(n8_x   / 256), 256, 0, stream>>>(x,   xbf,   n8_x);
  moe_cvt_bf16<<<(int)(n8_w13 / 256), 256, 0, stream>>>(w13, w13bf, n8_w13);
  moe_cvt_bf16<<<(int)(n8_w2  / 256), 256, 0, stream>>>(w2,  w2bf,  n8_w2);

  moe_gemm1  <<<dim3(IDIM / NT, T_TOK / MT, NEXP), 256, 0, stream>>>(xbf, w13bf, cnt, padoff, list, act);
  moe_gemm2  <<<dim3(HDIM / NT, T_TOK / MT, NEXP), 256, 0, stream>>>(act, w2bf, cnt, padoff, y);
  moe_combine<<<(T_TOK * HDIM / 4) / 256, 256, 0, stream>>>(y, topk_id, topk_w, pos, padoff, out);
}
